// TraceableFbank_9070970929585
// MI455X (gfx1250) — matrix operand order repaired and checked by reading, not compiled, not run on hardware
//
#include <hip/hip_runtime.h>
#include <hip/hip_bf16.h>
#include <math.h>

#ifndef __has_builtin
#define __has_builtin(x) 0
#endif
#if __has_builtin(__builtin_amdgcn_tensor_load_to_lds)
#define HAVE_TDM 1
#else
#define HAVE_TDM 0
#endif

// ---- problem constants (from reference) ----
constexpr int FRAME_LEN   = 400;
constexpr int SHIFT       = 160;
constexpr int PADW        = 512;
constexpr int NBINS       = 257;          // 512/2 + 1
constexpr int NMEL        = 80;
constexpr int BATCH       = 32;
constexpr int TSAMP       = 480000;
constexpr int FRAMES_PER  = (TSAMP - FRAME_LEN) / SHIFT + 1;  // 2998
constexpr int BF          = BATCH * FRAMES_PER;               // 95936
// ---- tiling ----
constexpr int ROWS        = 128;                              // frames per block (8 waves x 16)
constexpr int NBLK        = (BF + ROWS - 1) / ROWS;           // 750 (last block: 64 rows)
constexpr int NPAD  = 272;   // 17 N-tiles of 16 cover 257 bins (rows 257..271 zeroed)
constexpr int KP2   = 288;   // mel K padded to 9 k-steps of 32
constexpr int FROW  = 520;   // LDS frame row stride (f16): optimal LDS banking
constexpr int PROW  = 296;   // LDS power row stride (f16): optimal LDS banking
constexpr int BROW  = 520;   // LDS staged-B row stride (f16) == TDM pad 4 dw / 256 dw
constexpr float PREEMPH = 0.97f;
constexpr float EPS     = 1e-6f;

typedef __attribute__((ext_vector_type(16))) _Float16 v16h;
typedef __attribute__((ext_vector_type(8)))  _Float16 v8h;
typedef __attribute__((ext_vector_type(8)))  float    v8f;
typedef __attribute__((ext_vector_type(4)))  unsigned int u32x4;
typedef __attribute__((ext_vector_type(8)))  int      i32x8;
typedef __attribute__((ext_vector_type(4)))  int      i32x4;

// ---------------------------------------------------------------------------
// Prep: f32 -> f16 copies of DFT (padded [272][512], rows >=257 zeroed) and
// mel weights (padded [80][288], cols >= 257 zeroed) in d_ws (~600 KB).
// ---------------------------------------------------------------------------
__global__ void fbank_prep_f16(const float* __restrict__ dr,
                               const float* __restrict__ di,
                               const float* __restrict__ mw,
                               _Float16* __restrict__ dR,
                               _Float16* __restrict__ dI,
                               _Float16* __restrict__ M) {
    int idx = blockIdx.x * 256 + threadIdx.x;
    if (idx < NPAD * PADW) {
        int n = idx >> 9;
        int k = idx & (PADW - 1);
        float r = 0.f, im = 0.f;
        if (n < NBINS) { r = dr[n * PADW + k]; im = di[n * PADW + k]; }
        dR[idx] = (_Float16)r;
        dI[idx] = (_Float16)im;
    }
    if (idx < NMEL * KP2) {
        int n = idx / KP2;
        int k = idx - n * KP2;
        float v = (k < NBINS) ? mw[n * NBINS + k] : 0.f;
        M[idx] = (_Float16)v;
    }
}

#if HAVE_TDM
// ---------------------------------------------------------------------------
// TDM descriptor (CDNA5 ISA): copy one B tile
//   3D tensor: dim0 = 512 f16 (k), dim1 = 16 rows (stride 512),
//              dim2 = 2 matrices (stride NPAD*512: dI follows dR in ws)
//   LDS pad: 4 DWORDs after every 256 DWORDs -> row stride 520 f16 (= BROW),
//   i.e. the bank-conflict-free layout, produced by the DMA engine itself.
// ---------------------------------------------------------------------------
__device__ __forceinline__ void tdm_load_Btile(const _Float16* gsrc, unsigned lds_off) {
    unsigned long long ga = (unsigned long long)(uintptr_t)gsrc;
    u32x4 g0;
    g0.x = 1u;                                     // count=1 (valid user D#)
    g0.y = lds_off;                                // lds_addr (bytes)
    g0.z = (unsigned)ga;                           // global_addr[31:0]
    g0.w = (unsigned)((ga >> 32) & 0x01FFFFFFu)    // global_addr[56:32]
         | (2u << 30);                             // type = 2 ("image")
    i32x8 g1;
    g1[0] = (1 << 16)                              // data_size = 2 bytes
          | (1 << 20)                              // pad_enable
          | (7 << 22)                              // pad_interval: 256 DWORDs
          | (3 << 25);                             // pad_amount : 4 DWORDs
    g1[1] = (int)(512u << 16);                     // tensor_dim0[15:0]=512
    g1[2] = (int)(16u  << 16);                     // dim0 hi=0 ; tensor_dim1[15:0]=16
    g1[3] = (int)(512u << 16);                     // dim1 hi=0 ; tile_dim0=512
    g1[4] = (int)(16u | (2u << 16));               // tile_dim1=16, tile_dim2=2
    g1[5] = 512;                                   // tensor_dim0_stride[31:0]
    g1[6] = (int)(0x2000u << 16);                  // stride0 hi=0; stride1[15:0] (139264&0xffff)
    g1[7] = 0x2;                                   // stride1[47:16] (139264>>16)
    i32x4 g2 = (i32x4){2, 0, 0, 0};                // tensor_dim2=2; dim3/stride2/tile_dim3=0
    i32x4 g3 = (i32x4){0, 0, 0, 0};
#if defined(__clang_major__) && (__clang_major__ >= 23)
    i32x8 g4 = (i32x8){0, 0, 0, 0, 0, 0, 0, 0};
    __builtin_amdgcn_tensor_load_to_lds(g0, g1, g2, g3, g4, 0);
#else
    __builtin_amdgcn_tensor_load_to_lds(g0, g1, g2, g3, 0);
#endif
}
#endif  // HAVE_TDM

// ---------------------------------------------------------------------------
// Fused fbank:
//   block = 256 threads (8 wave32), 128 frames; wave w owns rows 16w..16w+15.
//   - B tile (real+imag, 33 KB) staged into double-buffered LDS once per block
//     per N-tile via the Tensor Data Mover (or manual staging fallback),
//     shared by all 8 waves -> 8x L2 traffic reduction; the DMA for nt+1 is
//     in flight while the WMMA burst for nt runs.
//   - The wave's A-tile (16 k-steps) is gathered from LDS ONCE into 128 VGPRs
//     and reused across all 17 N-tiles -> inner loop is 2 ds_load_b128 per
//     WMMA, the LDS-bandwidth floor for 32 B B-fragments.
// ---------------------------------------------------------------------------
__global__ void __launch_bounds__(256)
fbank_wmma(const float* __restrict__ wav,
           const float* __restrict__ window,
           const _Float16* __restrict__ dR,
           const _Float16* __restrict__ dI,
           const _Float16* __restrict__ melw,
           float* __restrict__ out) {
    __shared__ __align__(16) _Float16 sF[ROWS * FROW];        // frames (wave-private rows)
    __shared__ __align__(16) _Float16 sP[ROWS * PROW];        // power  (wave-private rows)
    __shared__ __align__(16) _Float16 sB[2 * 2 * 16 * BROW];  // staged B: [buf][R/I][row][k]

    const int tid  = threadIdx.x;
    const int lane = tid & 31;
    const int wave = tid >> 5;
    const int rowbase = wave * 16;                 // this wave's M-tile rows
    const long gfbase = (long)blockIdx.x * ROWS + rowbase;

    // ---- stage 0: zero this wave's power rows (pads cols >= 257 for mel K) ----
    {
        _Float16* sPw = sP + rowbase * PROW;
        const v8h z = {};
        for (int i = lane * 8; i < 16 * PROW; i += 32 * 8) *(v8h*)(sPw + i) = z;
    }

    // ---- stage 1: preprocess this wave's 16 frames into sF ----
    for (int rr = 0; rr < 16; ++rr) {
        const int r = rowbase + rr;
        const long gf = (long)blockIdx.x * ROWS + r;
        if (gf < BF) {
            const int b  = (int)(gf / FRAMES_PER);
            const int fr = (int)(gf - (long)b * FRAMES_PER);
            const float* src = wav + (size_t)b * TSAMP + (size_t)fr * SHIFT;
            float s = 0.f;
            for (int i = lane; i < FRAME_LEN; i += 32) s += src[i];
            #pragma unroll
            for (int off = 16; off; off >>= 1) s += __shfl_xor(s, off, 32);
            const float mean = s * (1.0f / (float)FRAME_LEN);
            for (int i = lane; i < PADW; i += 32) {
                float v = 0.f;
                if (i < FRAME_LEN) {
                    float x  = src[i];
                    float xp = (i == 0) ? x : src[i - 1];
                    v = ((x - mean) - PREEMPH * (xp - mean)) * window[i];
                }
                sF[r * FROW + i] = (_Float16)v;
            }
        } else {
            for (int i = lane; i < PADW; i += 32) sF[r * FROW + i] = (_Float16)0.f;
        }
    }

    // Per-lane fragment geometry (CDNA5 16-bit WMMA, wave32):
    //  A 16x32: lanes 0-15 (M=lane)   K kb+0..15 (order used here)
    //           lanes 16-31 (M=ln-16) K kb+16..31
    //  B 32x16: lane -> col n=lane&15; lanes<16 K kb+0..15, lanes>=16 K kb+16..31 (same order as A; not the ISA table order)
    const int nrow = lane & 15;
    const int sel  = lane >> 4;
    const int aoff = sel * 8;
    const int boff = sel * 16;

    // ---- stage 2: DFT; B via double-buffered LDS; A resident in 128 VGPRs ----
    constexpr unsigned BUFBYTES = 2u * 16u * (unsigned)BROW * 2u;   // 33280
#if HAVE_TDM
    const unsigned sB_lds = (unsigned)(uintptr_t)(&sB[0]);
    if (wave == 0) tdm_load_Btile(dR, sB_lds);            // DMA tile nt=0 -> buf 0
#else
    v8h stg[8];
    #pragma unroll
    for (int j = 0; j < 8; ++j) {
        const int c = tid + 256 * j;
        const int m = c >> 10, row = (c >> 6) & 15, kc = (c & 63) * 8;
        stg[j] = *(const v8h*)((m ? dI : dR) + (size_t)row * PADW + kc);
    }
#endif

    // gather this wave's 16 A-fragments once (reused for all 17 N-tiles)
    v16h afr[16];
    {
        const _Float16* arow = sF + (rowbase + nrow) * FROW;
        #pragma unroll
        for (int ks = 0; ks < 16; ++ks) {
            v8h alo = *(const v8h*)(arow + ks * 32 + boff);
            v8h ahi = *(const v8h*)(arow + ks * 32 + boff + 8);
            afr[ks] = __builtin_shufflevector(alo, ahi,
                        0,1,2,3,4,5,6,7,8,9,10,11,12,13,14,15);
        }
    }

    for (int nt = 0; nt < 17; ++nt) {
        const int buf = nt & 1;
#if HAVE_TDM
        if (wave == 0) __builtin_amdgcn_s_wait_tensorcnt(0);  // DMA for buf done
        __syncthreads();   // publish buf; everyone finished reading buf^1 (nt-1)
        if (nt < 16 && wave == 0)                              // DMA nt+1 into buf^1,
            tdm_load_Btile(dR + (size_t)(nt + 1) * 16 * PADW,  // overlaps WMMAs below
                           sB_lds + (unsigned)(buf ^ 1) * BUFBYTES);
#else
        __syncthreads();   // sB[buf] free (compute from it finished 2 iters ago)
        #pragma unroll
        for (int j = 0; j < 8; ++j) {
            const int c = tid + 256 * j;
            const int m = c >> 10, row = (c >> 6) & 15, kc = (c & 63) * 8;
            *(v8h*)(sB + (((buf << 1) + m) * 16 + row) * BROW + kc) = stg[j];
        }
        __syncthreads();   // sB[buf] visible to all waves
        if (nt < 16) {
            #pragma unroll
            for (int j = 0; j < 8; ++j) {
                const int c = tid + 256 * j;
                const int m = c >> 10, row = (c >> 6) & 15, kc = (c & 63) * 8;
                stg[j] = *(const v8h*)((m ? dI : dR) +
                                       (size_t)((nt + 1) * 16 + row) * PADW + kc);
            }
        }
#endif
        // 32 WMMAs per wave; only B comes from LDS (4 ds_load_b128 / 2 WMMAs)
        v8f accR = {}, accI = {};
        const _Float16* bRrow = sB + (((buf << 1) + 0) * 16 + nrow) * BROW;
        const _Float16* bIrow = sB + (((buf << 1) + 1) * 16 + nrow) * BROW;
        for (int ks = 0; ks < 16; ++ks) {
            const int kb = ks * 32;
            v8h rlo = *(const v8h*)(bRrow + kb + boff);
            v8h rhi = *(const v8h*)(bRrow + kb + boff + 8);
            v16h bR = __builtin_shufflevector(rlo, rhi,
                        0,1,2,3,4,5,6,7,8,9,10,11,12,13,14,15);
            v8h ilo = *(const v8h*)(bIrow + kb + boff);
            v8h ihi = *(const v8h*)(bIrow + kb + boff + 8);
            v16h bI = __builtin_shufflevector(ilo, ihi,
                        0,1,2,3,4,5,6,7,8,9,10,11,12,13,14,15);
            accR = __builtin_amdgcn_wmma_f32_16x16x32_f16(false, afr[ks], false, bR,
                                                          (short)0, accR, false, false);
            accI = __builtin_amdgcn_wmma_f32_16x16x32_f16(false, afr[ks], false, bI,
                                                          (short)0, accI, false, false);
        }
        // power -> sP (C/D layout: lanes<16 -> M=v, N=lane; lanes>=16 -> M=8+v)
        const int m0 = sel * 8;
        #pragma unroll
        for (int v = 0; v < 8; ++v) {
            float p = accR[v] * accR[v] + accI[v] * accI[v];
            sP[(rowbase + m0 + v) * PROW + nt * 16 + nrow] = (_Float16)p;
        }
    }

    // ---- stage 3: mel projection for this wave's M-tile (K=288 padded) + log ----
    // (sP rows read here were written by this same wave: same-wave DS ordering.)
    v16h pfr[9];
    {
        const _Float16* arow = sP + (rowbase + nrow) * PROW;
        #pragma unroll
        for (int ks = 0; ks < 9; ++ks) {
            v8h alo = *(const v8h*)(arow + ks * 32 + boff);
            v8h ahi = *(const v8h*)(arow + ks * 32 + boff + 8);
            pfr[ks] = __builtin_shufflevector(alo, ahi,
                        0,1,2,3,4,5,6,7,8,9,10,11,12,13,14,15);
        }
    }
    for (int nt = 0; nt < 5; ++nt) {
        v8f acc = {};
        const _Float16* bMrow = melw + (size_t)(nt * 16 + nrow) * KP2;
        for (int ks = 0; ks < 9; ++ks) {
            v16h bm = *(const v16h*)(bMrow + ks * 32 + boff);   // 32B aligned global
            acc = __builtin_amdgcn_wmma_f32_16x16x32_f16(false, pfr[ks], false, bm,
                                                         (short)0, acc, false, false);
        }
        const int m0 = sel * 8;
        #pragma unroll
        for (int v = 0; v < 8; ++v) {
            const long gf = gfbase + m0 + v;
            if (gf < BF) {
                float x = acc[v] + EPS;
                x = fmaxf(x, EPS);
                out[(size_t)gf * NMEL + nt * 16 + nrow] = __logf(x);
            }
        }
    }
}

// ---------------------------------------------------------------------------
extern "C" void kernel_launch(void* const* d_in, const int* in_sizes, int n_in,
                              void* d_out, int out_size, void* d_ws, size_t ws_size,
                              hipStream_t stream) {
    const float* wav    = (const float*)d_in[0];  // [32, 480000]
    const float* window = (const float*)d_in[1];  // [400]
    const float* dr     = (const float*)d_in[2];  // [257, 512]
    const float* di     = (const float*)d_in[3];  // [257, 512]
    const float* mw     = (const float*)d_in[4];  // [80, 257]
    float* out = (float*)d_out;                   // [32, 2998, 80] flat

    _Float16* ws  = (_Float16*)d_ws;
    _Float16* dR  = ws;                             // [272][512] f16
    _Float16* dI  = ws + (size_t)NPAD * PADW;       // [272][512] f16 (contiguous after dR!)
    _Float16* mel = ws + (size_t)2 * NPAD * PADW;   // [80][288]  f16
    // total ws use ~= 603 KB

    const int prepN = NPAD * PADW;                  // 139264
    fbank_prep_f16<<<(prepN + 255) / 256, 256, 0, stream>>>(dr, di, mw, dR, dI, mel);
    fbank_wmma<<<NBLK, 256, 0, stream>>>(wav, window, dR, dI, mel, out);
}
